// SelfAttention3D_74981539053694
// MI455X (gfx1250) — matrix operand order repaired and checked by reading, not compiled, not run on hardware
//
#include <hip/hip_runtime.h>

// ---------------------------------------------------------------------------
// SelfAttention3D for MI455X (gfx1250): all GEMMs via v_wmma_f32_16x16x32_f16,
// staging via TDM tensor_load_to_lds (6-arg builtin) and
// global_load_async_to_lds_b128 (ASYNCcnt), sync-copy fallbacks guarded.
// Pipeline: prep weights (f16 + head-summed Wo) -> 3 projections -> fused
// attention per (n,s) -> output GEMM (+bias) to f32.
// ---------------------------------------------------------------------------

typedef __attribute__((ext_vector_type(16))) _Float16 v16h;
typedef __attribute__((ext_vector_type(8)))  _Float16 v8h;
typedef __attribute__((ext_vector_type(8)))  float    v8f;
typedef __attribute__((ext_vector_type(4)))  unsigned int v4u;
typedef __attribute__((ext_vector_type(4)))  int      v4i;
typedef __attribute__((ext_vector_type(8)))  int      v8i;

#define DEV __device__ __forceinline__
#define AS1 __attribute__((address_space(1)))
#define AS3 __attribute__((address_space(3)))

static constexpr int NB = 4, SSE = 64;
static constexpr int M_TOT = NB * 128 * SSE; // 32768 flattened rows

DEV v16h cat16(v8h lo, v8h hi) {
    return __builtin_shufflevector(lo, hi, 0,1,2,3,4,5,6,7,8,9,10,11,12,13,14,15);
}

DEV v8f wmma16(v16h a, v16h b, v8f c) {
    // D = A(16x32 f16) * B(32x16 f16) + C(16x16 f32)
    return __builtin_amdgcn_wmma_f32_16x16x32_f16(
        false, a, false, b, (short)0, c, false, false);
}

// --- CDNA5 async copy global->LDS (ASYNCcnt path), sync fallback -----------
DEV void async_b128(const _Float16* g, _Float16* l) {
#if defined(__gfx1250__) && __has_builtin(__builtin_amdgcn_global_load_async_to_lds_b128)
    __builtin_amdgcn_global_load_async_to_lds_b128(
        (AS1 v4i*)g, (AS3 v4i*)l, 0, 0);
#else
    *(v8h*)l = *(const v8h*)g;
#endif
}

DEV void wait_async0() {
#if defined(__gfx1250__)
#if __has_builtin(__builtin_amdgcn_s_wait_asynccnt)
    __builtin_amdgcn_s_wait_asynccnt(0);
#else
    asm volatile("s_wait_asynccnt 0x0" ::: "memory");
#endif
#endif
}
DEV void wait_async2() {
#if defined(__gfx1250__)
#if __has_builtin(__builtin_amdgcn_s_wait_asynccnt)
    __builtin_amdgcn_s_wait_asynccnt(2);
#else
    asm volatile("s_wait_asynccnt 0x2" ::: "memory");
#endif
#endif
}
DEV void wait_tensor0() {
#if defined(__gfx1250__)
#if __has_builtin(__builtin_amdgcn_s_wait_tensorcnt)
    __builtin_amdgcn_s_wait_tensorcnt(0);
#else
    asm volatile("s_wait_tensorcnt 0x0" ::: "memory");
#endif
#endif
}

DEV unsigned lds_byte_off(const void* p) {
    return (unsigned)(unsigned long long)(AS3 const void*)p;
}

// ---------------------------------------------------------------------------
// Kernel 1: convert Wq/Wk/Wv to f16; fold the 8 head-copies of Wo into one
// 256x256 f16 matrix (attn output is identical across heads).
// ---------------------------------------------------------------------------
__global__ __launch_bounds__(256) void prep_weights(
        const float* __restrict__ Wq, const float* __restrict__ Wk,
        const float* __restrict__ Wv, const float* __restrict__ Wo,
        _Float16* __restrict__ Wq16, _Float16* __restrict__ Wk16,
        _Float16* __restrict__ Wv16, _Float16* __restrict__ Wo16) {
    int idx = blockIdx.x * 256 + threadIdx.x;   // 0..65535 over (d,e)
    Wq16[idx] = (_Float16)Wq[idx];
    Wk16[idx] = (_Float16)Wk[idx];
    Wv16[idx] = (_Float16)Wv[idx];
    float s = 0.0f;
#pragma unroll
    for (int h = 0; h < 8; ++h) s += Wo[(size_t)h * 65536 + idx];
    Wo16[idx] = (_Float16)s;
}

// ---------------------------------------------------------------------------
// Kernel 2/5: C[M,256] = X[M,256] @ W[256,256] (+bias). 128x128 tile/block,
// 8 waves x 16x128 stripe. Double-buffered k-loop: stage k+1 (async for f16
// input) while computing k; retire older async group with s_wait_asynccnt 2.
// ---------------------------------------------------------------------------
template <bool IN_F32, bool OUT_F32, bool BIAS>
__global__ __launch_bounds__(256) void gemm_256(
        const void* __restrict__ Xv, const _Float16* __restrict__ W,
        void* __restrict__ Ov, const float* __restrict__ bias) {
    constexpr int LDA = 48, LDB = 48;                   // 96B rows: 16B-aligned
    __shared__ __align__(16) _Float16 lA[2][128 * LDA];
    __shared__ __align__(16) _Float16 lB[2][128 * LDB];

    const int tid = threadIdx.x;
    const int w = tid >> 5, lane = tid & 31, l15 = lane & 15, hi = lane >> 4;
    const int m0 = blockIdx.x * 128, n0 = blockIdx.y * 128;
    const float*    Xf = (const float*)Xv;
    const _Float16* Xh = (const _Float16*)Xv;

    v8f c[8];
#pragma unroll
    for (int j = 0; j < 8; ++j)
#pragma unroll
        for (int r = 0; r < 8; ++r) c[j][r] = 0.0f;

    auto stage = [&](int buf, int k0) {
        if constexpr (IN_F32) {     // f32 input: load+convert (sync)
            int col = tid & 31, rb = tid >> 5;
#pragma unroll
            for (int i = 0; i < 16; ++i) {
                int row = rb + 8 * i;
                lA[buf][row * LDA + col] =
                    (_Float16)Xf[(size_t)(m0 + row) * 256 + k0 + col];
            }
        } else {                    // f16 input: async DMA to LDS, b128 chunks
#pragma unroll
            for (int ch = 0; ch < 2; ++ch) {
                int cidx = tid + 256 * ch;          // 512 chunks = 128r x 4
                int row = cidx >> 2, q = cidx & 3;
                async_b128(Xh + (size_t)(m0 + row) * 256 + k0 + 8 * q,
                           &lA[buf][row * LDA + 8 * q]);
            }
        }
        { // B tile stored transposed lB[n][k] for contiguous operand reads
            int nc = tid & 127, kb = tid >> 7;
#pragma unroll
            for (int i = 0; i < 16; ++i) {
                int kr = kb + 2 * i;
                lB[buf][nc * LDB + kr] = W[(size_t)(k0 + kr) * 256 + n0 + nc];
            }
        }
    };

    stage(0, 0);
    for (int it = 0; it < 8; ++it) {
        if (it < 7) stage((it + 1) & 1, 32 * (it + 1));
        if constexpr (!IN_F32) {
            if (it < 7) wait_async2();   // retire stage(it), stage(it+1) flies
            else        wait_async0();
        }
        __syncthreads();
        const int buf = it & 1;
        // A operand: lane l15 = M row; k = (j&7) + 8*hi + 16*(j>>3)
        const _Float16* ap = &lA[buf][(16 * w + l15) * LDA + 8 * hi];
        v16h a = cat16(*(const v8h*)ap, *(const v8h*)(ap + 16));
#pragma unroll
        for (int j = 0; j < 8; ++j) {
            // B operand: lane l15 = N col; k = (t&7) + 8*hi + 16*(t>>3)
            const _Float16* bp = &lB[buf][(16 * j + l15) * LDB + 8 * hi];
            v16h b = cat16(*(const v8h*)bp, *(const v8h*)(bp + 16));
            c[j] = wmma16(a, b, c[j]);
        }
        __syncthreads();   // safe to overwrite this buffer two stages later
    }

    float*    Of = (float*)Ov;
    _Float16* Oh = (_Float16*)Ov;
#pragma unroll
    for (int j = 0; j < 8; ++j)
#pragma unroll
        for (int r = 0; r < 8; ++r) {
            int row = m0 + 16 * w + r + 8 * hi;   // C layout: M = r + 8*hi
            int col = n0 + 16 * j + l15;          //           N = lane&15
            float v = c[j][r];
            if (BIAS) v += bias[col];
            if (OUT_F32) Of[(size_t)row * 256 + col] = v;
            else         Oh[(size_t)row * 256 + col] = (_Float16)v;
        }
}

// ---------------------------------------------------------------------------
// Kernel 3: fused attention for one (n,s): E = Q K^T (128x128, K=256),
// masked softmax over k, O = A V (128x256, K=128).
// K tile -> LDS via TDM (2D descriptor, LDS row padding 512B+16B) when the
// tensor builtin exists, else async-to-LDS. V is transposed into LDS via
// VGPRs. Attn weights overwrite the dead Ks region after a barrier.
// ---------------------------------------------------------------------------
__global__ __launch_bounds__(256) void attn_fused(
        const _Float16* __restrict__ Qp, const _Float16* __restrict__ Kp,
        const _Float16* __restrict__ Vp, const int* __restrict__ mask,
        _Float16* __restrict__ Op) {
    constexpr int KS_LD = 264, VT_LD = 136, AS_LD = 136; // rows 16B-aligned
    __shared__ __align__(16) _Float16 smem[128 * KS_LD + 256 * VT_LD]; // 137 KB
    _Float16* Ks = smem;
    _Float16* Vt = smem + 128 * KS_LD;
    _Float16* As = smem; // aliases Ks (reused after barrier)

    const int tid = threadIdx.x;
    const int w = tid >> 5, lane = tid & 31, l15 = lane & 15, hi = lane >> 4;
    const int n = blockIdx.x >> 6, s = blockIdx.x & 63;

    // ---- stage K tile: 128 rows x 512B, row stride 32KB, padded LDS rows ---
#if defined(__gfx1250__) && __has_builtin(__builtin_amdgcn_tensor_load_to_lds)
    if (w == 0) {
        unsigned long long ga = (unsigned long long)
            (Kp + ((size_t)(n * 128) * 64 + s) * 256);
        v4u g0; v8i g1;
        v4i g2 = {0, 0, 0, 0}, g3 = {0, 0, 0, 0};
        v8i g4 = {0, 0, 0, 0, 0, 0, 0, 0};
        g0[0] = 1u;                                  // count=1, user mode
        g0[1] = lds_byte_off(Ks);                    // lds_addr
        g0[2] = (unsigned)(ga & 0xFFFFFFFFu);        // global_addr[31:0]
        g0[3] = (unsigned)((ga >> 32) & 0x01FFFFFFu) // global_addr[56:32]
                | 0x80000000u;                       // type=2 (image)
        g1[0] = (1 << 16)       // data_size = 2B
              | (1 << 20)       // pad_enable
              | (6 << 22)       // pad_interval: 128 DWORDs = 512B (one row)
              | (3 << 25);      // pad_amount: 4 DWORDs = 16B (KS_LD-256 halves)
        g1[1] = (int)(256u << 16);   // tensor_dim0 = 256 elements
        g1[2] = (int)(128u << 16);   // tensor_dim1 = 128
        g1[3] = (int)(256u << 16);   // tile_dim0 = 256
        g1[4] = 128;                 // tile_dim1 = 128 (tile_dim2 = 0)
        g1[5] = 16384;               // tensor_dim0_stride = 64*256 elements
        g1[6] = 0; g1[7] = 0;        // dim1 stride unused (2D)
        __builtin_amdgcn_tensor_load_to_lds(g0, g1, g2, g3, g4, 0);
    }
    wait_tensor0();
#else
    {
#pragma unroll
        for (int ch = 0; ch < 16; ++ch) {
            int cidx = tid + 256 * ch;          // 4096 chunks = 128r x 32
            int k = cidx >> 5, q = cidx & 31;
            size_t gb = ((size_t)(n * 128 + k) * 64 + s) * 256;
            async_b128(Kp + gb + 8 * q, &Ks[k * KS_LD + 8 * q]);
        }
        wait_async0();
    }
#endif

    // ---- stage V transposed: Vt[d][k] (b128 global reads, scatter stores) --
    {
        int c8 = (tid & 31) * 8, kb = tid >> 5;
#pragma unroll
        for (int it = 0; it < 16; ++it) {
            int k = kb + 8 * it;
            size_t gb = ((size_t)(n * 128 + k) * 64 + s) * 256;
            v8h vx = *(const v8h*)(Vp + gb + c8);
#pragma unroll
            for (int u = 0; u < 8; ++u) Vt[(c8 + u) * VT_LD + k] = vx[u];
        }
    }
    __syncthreads();

    // ---- stage 1: E = Q * K^T, wave w owns rows 16w..16w+15 ----
    v8f c[8];
#pragma unroll
    for (int j = 0; j < 8; ++j)
#pragma unroll
        for (int r = 0; r < 8; ++r) c[j][r] = 0.0f;

    const _Float16* Qrow = Qp + ((size_t)(n * 128 + 16 * w + l15) * 64 + s) * 256;
#pragma unroll
    for (int d0 = 0; d0 < 8; ++d0) {
        const _Float16* ap = Qrow + 32 * d0 + 8 * hi;
        v16h a = cat16(*(const v8h*)ap, *(const v8h*)(ap + 16));
#pragma unroll
        for (int j = 0; j < 8; ++j) {
            // B = K^T: B[d][k'] = K[k'][d]; lane l15 = column k' (row of Ks)
            const _Float16* bp = &Ks[(16 * j + l15) * KS_LD + 32 * d0 + 8 * hi];
            v16h b = cat16(*(const v8h*)bp, *(const v8h*)(bp + 16));
            c[j] = wmma16(a, b, c[j]);
        }
    }

    // ---- masked softmax over k; rows live across 16-lane half-groups ----
    const int qrow0 = 16 * w + 8 * hi;
    float mx[8], sm[8];
#pragma unroll
    for (int r = 0; r < 8; ++r) mx[r] = -3.0e38f;
#pragma unroll
    for (int j = 0; j < 8; ++j)
#pragma unroll
        for (int r = 0; r < 8; ++r) {
            int q = qrow0 + r, k = 16 * j + l15;
            int mk = mask[(size_t)n * 16384 + (size_t)q * 128 + k];
            float x = mk ? c[j][r] * 0.0625f : -6.25e18f; // (-1e20)/sqrt(256)
            c[j][r] = x;
            mx[r] = fmaxf(mx[r], x);
        }
#pragma unroll
    for (int r = 0; r < 8; ++r) {
        float m = mx[r];
        m = fmaxf(m, __shfl_xor(m, 1, 32));
        m = fmaxf(m, __shfl_xor(m, 2, 32));
        m = fmaxf(m, __shfl_xor(m, 4, 32));
        m = fmaxf(m, __shfl_xor(m, 8, 32));
        mx[r] = m;
        sm[r] = 0.0f;
    }
#pragma unroll
    for (int j = 0; j < 8; ++j)
#pragma unroll
        for (int r = 0; r < 8; ++r) {
            float p = __expf(c[j][r] - mx[r]);
            c[j][r] = p;
            sm[r] += p;
        }
#pragma unroll
    for (int r = 0; r < 8; ++r) {
        float t = sm[r];
        t += __shfl_xor(t, 1, 32);
        t += __shfl_xor(t, 2, 32);
        t += __shfl_xor(t, 4, 32);
        t += __shfl_xor(t, 8, 32);
        sm[r] = 1.0f / t;
    }

    __syncthreads(); // all waves done reading Ks; safe to overwrite with As
#pragma unroll
    for (int j = 0; j < 8; ++j)
#pragma unroll
        for (int r = 0; r < 8; ++r)
            As[(qrow0 + r) * AS_LD + 16 * j + l15] = (_Float16)(c[j][r] * sm[r]);
    // each wave reads back only rows it wrote -> per-wave DS ordering suffices

    // ---- stage 2: O = A * V (K = 128), D split in two halves of 128 ----
#pragma unroll
    for (int half = 0; half < 2; ++half) {
        v8f c2[8];
#pragma unroll
        for (int j = 0; j < 8; ++j)
#pragma unroll
            for (int r = 0; r < 8; ++r) c2[j][r] = 0.0f;
#pragma unroll
        for (int ks = 0; ks < 4; ++ks) {
            const _Float16* ap = &As[(16 * w + l15) * AS_LD + 32 * ks + 8 * hi];
            v16h a = cat16(*(const v8h*)ap, *(const v8h*)(ap + 16));
#pragma unroll
            for (int j = 0; j < 8; ++j) {
                int d = 128 * half + 16 * j + l15;
                const _Float16* bp = &Vt[d * VT_LD + 32 * ks + 8 * hi];
                v16h b = cat16(*(const v8h*)bp, *(const v8h*)(bp + 16));
                c2[j] = wmma16(a, b, c2[j]);
            }
        }
#pragma unroll
        for (int j = 0; j < 8; ++j)
#pragma unroll
            for (int r = 0; r < 8; ++r) {
                int q = 16 * w + r + 8 * hi;
                int d = 128 * half + 16 * j + l15;
                Op[((size_t)(n * 128 + q) * 64 + s) * 256 + d] = (_Float16)c2[j][r];
            }
    }
}

// ---------------------------------------------------------------------------
// Launch. Inputs: values, keys, query, mask, Wq, Wk, Wv, Wo, bo.
// Workspace: 4 f16 weight mats (512 KB) + Qp/Kp/Vp/Op f16 (16 MB each).
// ---------------------------------------------------------------------------
extern "C" void kernel_launch(void* const* d_in, const int* in_sizes, int n_in,
                              void* d_out, int out_size, void* d_ws, size_t ws_size,
                              hipStream_t stream) {
    (void)in_sizes; (void)n_in; (void)out_size; (void)ws_size;
    const float* values = (const float*)d_in[0];
    const float* keys   = (const float*)d_in[1];
    const float* query  = (const float*)d_in[2];
    const int*   mask   = (const int*)d_in[3];
    const float* Wq = (const float*)d_in[4];
    const float* Wk = (const float*)d_in[5];
    const float* Wv = (const float*)d_in[6];
    const float* Wo = (const float*)d_in[7];
    const float* bo = (const float*)d_in[8];

    char* ws = (char*)d_ws;
    const size_t WSZ = (size_t)256 * 256 * sizeof(_Float16); // 128 KB
    _Float16* Wq16 = (_Float16*)(ws);
    _Float16* Wk16 = (_Float16*)(ws + WSZ);
    _Float16* Wv16 = (_Float16*)(ws + 2 * WSZ);
    _Float16* Wo16 = (_Float16*)(ws + 3 * WSZ);
    const size_t base = 4 * WSZ;
    const size_t PSZ = (size_t)M_TOT * 256 * sizeof(_Float16); // 16 MB
    _Float16* Qp = (_Float16*)(ws + base);
    _Float16* Kp = (_Float16*)(ws + base + PSZ);
    _Float16* Vp = (_Float16*)(ws + base + 2 * PSZ);
    _Float16* Op = (_Float16*)(ws + base + 3 * PSZ);

    prep_weights<<<256, 256, 0, stream>>>(Wq, Wk, Wv, Wo, Wq16, Wk16, Wv16, Wo16);

    dim3 g(M_TOT / 128, 2);
    gemm_256<true,  false, false><<<g, 256, 0, stream>>>(query,  Wq16, Qp, nullptr);
    gemm_256<true,  false, false><<<g, 256, 0, stream>>>(keys,   Wk16, Kp, nullptr);
    gemm_256<true,  false, false><<<g, 256, 0, stream>>>(values, Wv16, Vp, nullptr);

    attn_fused<<<NB * SSE, 256, 0, stream>>>(Qp, Kp, Vp, mask, Op);

    gemm_256<false, true, true><<<g, 256, 0, stream>>>(Op, Wo16, d_out, bo);
}
